// UpdateRule_60301340836020
// MI455X (gfx1250) — matrix operand order repaired and checked by reading, not compiled, not run on hardware
//
#include <hip/hip_runtime.h>

#define HID    80
#define XW     82
#define SDEG   9
#define EDIM   89
#define EP     96
#define QKVW   288
#define BNODES 4
#define ROWS   36     // BNODES * SDEG
#define RPAD   48     // 3 M-tiles of 16
#define MTI    3
#define KTI    3      // 96 / 32
#define NT_QKV 18     // 288 / 16
#define NT_OUT 6      // 96 / 16
#define PNB    128
#define FQSZ   (NT_QKV*KTI*256)   // uints per (layer,mha) qkv frag region
#define FOSZ   (NT_OUT*KTI*256)   // uints per (layer,mha) out frag region

typedef __attribute__((ext_vector_type(16))) _Float16 v16h;
typedef __attribute__((ext_vector_type(8)))  float    v8f;

union FragU { unsigned u[8]; v16h h; };

__device__ __forceinline__ unsigned pack2h(float a, float b) {
  union { _Float16 h[2]; unsigned u; } p;
  p.h[0] = (_Float16)a; p.h[1] = (_Float16)b;
  return p.u;
}

// ---------------------------------------------------------------------------
// Pre-swizzle weights into WMMA B fragments (f16), in contiguous k order (same as load_afrag).
// B (K x N) fragment, wave32: lane L -> col = L&15, kb = (L<16)?0:16,
// vgpr i holds K = kb+2i, kb+2i+1 (low half = lower K).
// qkv_w is [lm][f(267)][e(89)]; B[e][f] = qkv_w[f][e]. f padded per 96-section.
// out_w is [lm][f(89)][e(89)];  B[e][f] = out_w[f][e].
// ---------------------------------------------------------------------------
__global__ void prep_frags(const float* __restrict__ qkv_w,
                           const float* __restrict__ out_w,
                           unsigned* __restrict__ fragQ,
                           unsigned* __restrict__ fragO) {
  const int lane = threadIdx.x;
  const int bid  = blockIdx.x;
  const int col  = lane & 15;
  const int kb   = (lane < 16) ? 0 : 16;
  if (bid < 4 * NT_QKV * KTI) {                 // 216 qkv fragments
    int lm = bid / (NT_QKV * KTI);
    int rest = bid % (NT_QKV * KTI);
    int nt = rest / KTI, kt = rest % KTI;
    const float* Wsrc = qkv_w + (size_t)lm * 267 * EDIM;
    int fg  = nt * 16 + col;
    int sec = fg / EP, fi = fg % EP;
    unsigned* dst = fragQ + (((size_t)lm * NT_QKV * KTI + nt * KTI + kt) * 32 + lane) * 8;
    #pragma unroll
    for (int i = 0; i < 8; i++) {
      int K = kt * 32 + kb + 2 * i;
      float a = 0.f, b = 0.f;
      if (fi < EDIM) {
        int f = sec * EDIM + fi;
        if (K     < EDIM) a = Wsrc[f * EDIM + K];
        if (K + 1 < EDIM) b = Wsrc[f * EDIM + K + 1];
      }
      dst[i] = pack2h(a, b);
    }
  } else {                                       // 72 out-proj fragments
    int lin = bid - 4 * NT_QKV * KTI;
    int lm = lin / (NT_OUT * KTI);
    int rest = lin % (NT_OUT * KTI);
    int nt = rest / KTI, kt = rest % KTI;
    const float* Wsrc = out_w + (size_t)lm * EDIM * EDIM;
    int f = nt * 16 + col;
    unsigned* dst = fragO + (((size_t)lm * NT_OUT * KTI + nt * KTI + kt) * 32 + lane) * 8;
    #pragma unroll
    for (int i = 0; i < 8; i++) {
      int K = kt * 32 + kb + 2 * i;
      float a = 0.f, b = 0.f;
      if (f < EDIM) {
        if (K     < EDIM) a = Wsrc[f * EDIM + K];
        if (K + 1 < EDIM) b = Wsrc[f * EDIM + K + 1];
      }
      dst[i] = pack2h(a, b);
    }
  }
}

// ---------------------------------------------------------------------------
// h[n][c] = sum_e in[n][e] * W[e][c]   (W row-major Kd x 80)
// ---------------------------------------------------------------------------
__global__ __launch_bounds__(128) void node_gemm(const float* __restrict__ in,
                                                 const float* __restrict__ W,
                                                 float* __restrict__ out, int Kd) {
  __shared__ float row[96];
  const int n = blockIdx.x, tid = threadIdx.x;
  if (tid < Kd) row[tid] = in[(size_t)n * Kd + tid];
  __syncthreads();
  if (tid < HID) {
    float a = 0.f;
    for (int e = 0; e < Kd; e++) a += row[e] * W[e * HID + tid];
    out[(size_t)n * HID + tid] = a;
  }
}

// ---------------------------------------------------------------------------
// A-fragment loader: row-major LDS half matrix, stride EP halfs.
// lane L -> row = mt*16 + (L&15); kh = (L<16)?0:8 (uint units);
// vgpr i (0..7) : K pair 2i + 2*kh (contiguous, matches B)
// ---------------------------------------------------------------------------
__device__ __forceinline__ void load_afrag(FragU& a, const _Float16* base,
                                           int mt, int kt, int lane) {
  int row = mt * 16 + (lane & 15);
  int kh  = (lane < 16) ? 0 : 8;
  const unsigned* zp = (const unsigned*)base + (row * EP + kt * 32) / 2;
  #pragma unroll
  for (int i = 0; i < 4; i++) { a.u[i] = zp[kh + i]; a.u[4 + i] = zp[kh + 4 + i]; }
}

// ---------------------------------------------------------------------------
// Fused per-node attention conv: z build + 2x MHA (WMMA) + max-pool
// ---------------------------------------------------------------------------
__global__ __launch_bounds__(256) void conv_kernel(
    const float* __restrict__ h, const int* __restrict__ nbr,
    const unsigned* __restrict__ fq0, const unsigned* __restrict__ fo0,
    const unsigned* __restrict__ fq1, const unsigned* __restrict__ fo1,
    const float* __restrict__ qkvb0, const float* __restrict__ qkvb1,
    const float* __restrict__ outb0, const float* __restrict__ outb1,
    const float* __restrict__ bvec, float* __restrict__ uout) {
  __shared__ _Float16 zA[RPAD * EP];      //  9,216 B
  __shared__ _Float16 qkvS[RPAD * QKVW];  // 27,648 B
  __shared__ _Float16 oS[RPAD * EP];      //  9,216 B
  __shared__ float    zacc[RPAD * EP];    // 18,432 B  (total 64,512 B)

  const int tid = threadIdx.x;
  const int lane = tid & 31;
  const int wv = tid >> 5;
  const int nodeBase = blockIdx.x * BNODES;

  // phase 0: build z (gather + one-hot), init accumulators
  for (int idx = tid; idx < RPAD * EP; idx += 256) {
    int r = idx / EP, e = idx % EP;
    float v = 0.f;
    if (r < ROWS) {
      int b = r / SDEG, s = r % SDEG;
      int g = nbr[(nodeBase + b) * SDEG + s];
      if (e < HID) v = h[(size_t)g * HID + e];
      else if (e == HID + s) v = 1.f;
    }
    zA[idx] = (_Float16)v;
    oS[idx] = (_Float16)0.f;
    float bs = (r < ROWS && e < EDIM) ? (outb0[e] + outb1[e]) : 0.f;
    zacc[idx] = (r < ROWS) ? (v + bs) : 0.f;
  }
  __syncthreads();

  for (int m = 0; m < 2; m++) {
    const unsigned* fq = m ? fq1 : fq0;
    const unsigned* fo = m ? fo1 : fo0;
    const float* qb = m ? qkvb1 : qkvb0;

    // GEMM1: qkv = z @ Wqkv   (3 M-tiles x 18 N-tiles x 3 K-steps)
    for (int t = wv; t < MTI * NT_QKV; t += 8) {
      int mt = t / NT_QKV, nt = t % NT_QKV;
      v8f acc = {};
      for (int kt = 0; kt < KTI; kt++) {
        FragU a, b;
        load_afrag(a, zA, mt, kt, lane);
        const unsigned* bp = fq + ((size_t)(nt * KTI + kt) * 32 + lane) * 8;
        #pragma unroll
        for (int i = 0; i < 8; i++) b.u[i] = bp[i];
        acc = __builtin_amdgcn_wmma_f32_16x16x32_f16(false, a.h, false, b.h,
                                                     (short)0, acc, false, false);
      }
      int col = lane & 15;
      int fg = nt * 16 + col;
      int sec = fg / EP, fi = fg % EP;
      float bias = (fi < EDIM) ? qb[sec * EDIM + fi] : 0.f;
      int mb = (lane < 16) ? 0 : 8;
      #pragma unroll
      for (int i = 0; i < 8; i++)
        qkvS[(mt * 16 + i + mb) * QKVW + fg] = (_Float16)(acc[i] + bias);
    }
    __syncthreads();

    // attention core: one thread per (node, token) row; S=9, E=89
    if (tid < ROWS) {
      int b = tid / SDEG;
      const _Float16* qr = &qkvS[tid * QKVW];
      float sc[SDEG];
      float mx = -3.0e38f;
      #pragma unroll
      for (int t2 = 0; t2 < SDEG; t2++) {
        const _Float16* kr = &qkvS[(b * SDEG + t2) * QKVW + EP];
        float d = 0.f;
        for (int e = 0; e < EDIM; e++) d += (float)qr[e] * (float)kr[e];
        d *= 0.10599978800063601f;   // 1/sqrt(89)
        sc[t2] = d; mx = fmaxf(mx, d);
      }
      float den = 0.f;
      #pragma unroll
      for (int t2 = 0; t2 < SDEG; t2++) { sc[t2] = __expf(sc[t2] - mx); den += sc[t2]; }
      float inv = 1.f / den;
      for (int e = 0; e < EP; e++) {
        float o = 0.f;
        #pragma unroll
        for (int t2 = 0; t2 < SDEG; t2++)
          o += sc[t2] * (float)qkvS[(b * SDEG + t2) * QKVW + 2 * EP + e];
        oS[tid * EP + e] = (_Float16)(o * inv);
      }
    }
    __syncthreads();

    // GEMM2: zacc += o @ Wout  (3 M-tiles x 6 N-tiles x 3 K-steps)
    for (int t = wv; t < MTI * NT_OUT; t += 8) {
      int mt = t / NT_OUT, nt = t % NT_OUT;
      v8f acc = {};
      for (int kt = 0; kt < KTI; kt++) {
        FragU a, b;
        load_afrag(a, oS, mt, kt, lane);
        const unsigned* bp = fo + ((size_t)(nt * KTI + kt) * 32 + lane) * 8;
        #pragma unroll
        for (int i = 0; i < 8; i++) b.u[i] = bp[i];
        acc = __builtin_amdgcn_wmma_f32_16x16x32_f16(false, a.h, false, b.h,
                                                     (short)0, acc, false, false);
      }
      int col = lane & 15;
      int mb = (lane < 16) ? 0 : 8;
      #pragma unroll
      for (int i = 0; i < 8; i++) {
        int r = mt * 16 + i + mb;
        zacc[r * EP + nt * 16 + col] += acc[i];
      }
    }
    __syncthreads();
  }

  // max over tokens + bias
  for (int idx = tid; idx < BNODES * HID; idx += 256) {
    int b = idx / HID, c = idx % HID;
    float mx = zacc[(b * SDEG) * EP + c];
    #pragma unroll
    for (int s = 1; s < SDEG; s++) mx = fmaxf(mx, zacc[(b * SDEG + s) * EP + c]);
    uout[(size_t)(nodeBase + b) * HID + c] = mx + bvec[c];
  }
}

// ---------------------------------------------------------------------------
// PairNorm: deterministic two-stage reduction
// ---------------------------------------------------------------------------
__global__ __launch_bounds__(128) void pn_partial(const float* __restrict__ u,
                                                  float* __restrict__ part,
                                                  float* __restrict__ psq, int N) {
  __shared__ float sbuf[128];
  const int tid = threadIdx.x;
  float acc = 0.f, sq = 0.f;
  for (int r = blockIdx.x; r < N; r += gridDim.x) {
    if (tid < HID) {
      float v = u[(size_t)r * HID + tid];
      acc += v; sq += v * v;
    }
  }
  if (tid < HID) part[blockIdx.x * HID + tid] = acc;
  sbuf[tid] = (tid < HID) ? sq : 0.f;
  __syncthreads();
  for (int st = 64; st > 0; st >>= 1) {
    if (tid < st) sbuf[tid] += sbuf[tid + st];
    __syncthreads();
  }
  if (tid == 0) psq[blockIdx.x] = sbuf[0];
}

__global__ __launch_bounds__(128) void pn_final(const float* __restrict__ part,
                                                const float* __restrict__ psq,
                                                float* __restrict__ stats, int N) {
  __shared__ float meanS[HID];
  const int tid = threadIdx.x;
  if (tid < HID) {
    float a = 0.f;
    for (int b = 0; b < PNB; b++) a += part[b * HID + tid];
    meanS[tid] = a / (float)N;
    stats[tid] = meanS[tid];
  }
  __syncthreads();
  if (tid == 0) {
    float t2 = 0.f;
    for (int b = 0; b < PNB; b++) t2 += psq[b];
    float msq = 0.f;
    for (int c = 0; c < HID; c++) msq += meanS[c] * meanS[c];
    float val = fmaxf(t2 / (float)N - msq, 0.f);
    stats[HID] = rsqrtf(1e-5f + val);
  }
}

__global__ void pn_apply(const float* __restrict__ u, const float* __restrict__ stats,
                         float* __restrict__ y, int total) {
  int idx = blockIdx.x * blockDim.x + threadIdx.x;
  if (idx < total) {
    int c = idx % HID;
    float v = (u[idx] - stats[c]) * stats[HID];
    y[idx] = fmaxf(v, 0.f);
  }
}

// ---------------------------------------------------------------------------
// out = 0.5*(x[:, :80] + u), last column zeroed if |x+u| > 1 (pre-scale)
// ---------------------------------------------------------------------------
__global__ void final_combine(const float* __restrict__ x, const float* __restrict__ u,
                              float* __restrict__ out, int total) {
  int idx = blockIdx.x * blockDim.x + threadIdx.x;
  if (idx < total) {
    int n = idx / HID, c = idx % HID;
    float v = x[(size_t)n * XW + c] + u[idx];
    if (c == HID - 1 && (v > 1.f || v < -1.f)) v = 0.f;
    out[idx] = v * 0.5f;
  }
}

// ---------------------------------------------------------------------------
extern "C" void kernel_launch(void* const* d_in, const int* in_sizes, int n_in,
                              void* d_out, int out_size, void* d_ws, size_t ws_size,
                              hipStream_t stream) {
  const float* x    = (const float*)d_in[0];
  const float* W1   = (const float*)d_in[1];
  const float* b1   = (const float*)d_in[2];
  const float* W2   = (const float*)d_in[3];
  const float* b2   = (const float*)d_in[4];
  const float* qkvw = (const float*)d_in[5];
  const float* qkvb = (const float*)d_in[6];
  const float* outw = (const float*)d_in[7];
  const float* outb = (const float*)d_in[8];
  const int*   nbr  = (const int*)d_in[9];
  const int N = in_sizes[9] / SDEG;

  float* wsf = (float*)d_ws;
  size_t U_OFF = (size_t)N * HID;
  size_t V_OFF = 2 * (size_t)N * HID;
  size_t PART_OFF = 3 * (size_t)N * HID;
  size_t PSQ_OFF  = PART_OFF + (size_t)PNB * HID;
  size_t STAT_OFF = PSQ_OFF + PNB;
  size_t FRAG_OFF = STAT_OFF + 96;
  float* hbuf  = wsf;
  float* ubuf  = wsf + U_OFF;
  float* vbuf  = wsf + V_OFF;
  float* part  = wsf + PART_OFF;
  float* psq   = wsf + PSQ_OFF;
  float* stats = wsf + STAT_OFF;
  unsigned* fragQ = (unsigned*)(wsf + FRAG_OFF);
  unsigned* fragO = fragQ + 4 * FQSZ;

  prep_frags<<<4 * NT_QKV * KTI + 4 * NT_OUT * KTI, 32, 0, stream>>>(qkvw, outw, fragQ, fragO);

  // ---- layer 1 ----
  node_gemm<<<N, 128, 0, stream>>>(x, W1, hbuf, XW);
  conv_kernel<<<N / BNODES, 256, 0, stream>>>(hbuf, nbr,
      fragQ + 0 * FQSZ, fragO + 0 * FOSZ, fragQ + 1 * FQSZ, fragO + 1 * FOSZ,
      qkvb + 0 * 267, qkvb + 1 * 267, outb + 0 * EDIM, outb + 1 * EDIM, b1, ubuf);
  pn_partial<<<PNB, 128, 0, stream>>>(ubuf, part, psq, N);
  pn_final<<<1, 128, 0, stream>>>(part, psq, stats, N);
  pn_apply<<<(N * HID + 255) / 256, 256, 0, stream>>>(ubuf, stats, vbuf, N * HID);

  // ---- layer 2 ----
  node_gemm<<<N, 128, 0, stream>>>(vbuf, W2, hbuf, HID);
  conv_kernel<<<N / BNODES, 256, 0, stream>>>(hbuf, nbr,
      fragQ + 2 * FQSZ, fragO + 2 * FOSZ, fragQ + 3 * FQSZ, fragO + 3 * FOSZ,
      qkvb + 2 * 267, qkvb + 3 * 267, outb + 2 * EDIM, outb + 3 * EDIM, b2, ubuf);
  pn_partial<<<PNB, 128, 0, stream>>>(ubuf, part, psq, N);
  pn_final<<<1, 128, 0, stream>>>(part, psq, stats, N);
  pn_apply<<<(N * HID + 255) / 256, 256, 0, stream>>>(ubuf, stats, vbuf, N * HID);

  final_combine<<<(N * HID + 255) / 256, 256, 0, stream>>>(x, vbuf, (float*)d_out, N * HID);
}
